// MaskedMultiHeadAttention_30167850287346
// MI455X (gfx1250) — matrix operand order repaired and checked by reading, not compiled, not run on hardware
//
#include <hip/hip_runtime.h>
#include <hip/hip_bf16.h>

// ---- problem constants (from reference) ----
constexpr int Bc = 4, Lc = 1024, Sc = 1024, Ec = 512, Hc = 8, Dc = 64;

typedef _Float16 h16 __attribute__((ext_vector_type(16)));
typedef _Float16 h8  __attribute__((ext_vector_type(8)));
typedef float    f8  __attribute__((ext_vector_type(8)));
typedef unsigned int u32x4 __attribute__((ext_vector_type(4)));
typedef int          i32x4 __attribute__((ext_vector_type(4)));
typedef int          i32x8 __attribute__((ext_vector_type(8)));

#define F8_ZERO {0.f,0.f,0.f,0.f,0.f,0.f,0.f,0.f}

// =====================================================================
// WMMA operand loaders (ISA 7.12.2 layouts, wave32)
// A (16x32 f16): lanes 0-15 rows M=0..15 with K {0..7,16..23}; lanes 16-31
// same rows with K {8..15,24..31}.
// =====================================================================
__device__ __forceinline__ h16 load_a_f16(const _Float16* __restrict__ A,
                                          int lda, int row, int k0, int lane) {
    int grp = lane >> 4;
    const _Float16* p0 = A + (size_t)row * lda + k0 + grp * 8;
    h8 lo = *(const h8*)p0;
    h8 hi = *(const h8*)(p0 + 16);
    return __builtin_shufflevector(lo, hi, 0,1,2,3,4,5,6,7,8,9,10,11,12,13,14,15);
}

// Same layout but source is fp32; convert on the fly (WMMA operands are f16).
__device__ __forceinline__ h16 load_a_f32(const float* __restrict__ A,
                                          int lda, int row, int k0, int lane) {
    int grp = lane >> 4;
    const float* p0 = A + (size_t)row * lda + k0 + grp * 8;
    const float* p1 = p0 + 16;
    h16 a;
#pragma unroll
    for (int j = 0; j < 8; ++j) {
        a[j]     = (_Float16)p0[j];
        a[8 + j] = (_Float16)p1[j];
    }
    return a;
}

// B (32x16 f16), stored N-major (Bt[n*ldb + k]): lanes 0-15 hold K {0..7,16..23}
// lanes 16-31 hold K {8..15,24..31} (same as A); N = lane&15.
// Works for both global and LDS-derived pointers (addrspace inferred).
__device__ __forceinline__ h16 load_b_t(const _Float16* Bt,
                                        int ldb, int col, int k0, int lane) {
    int grp = lane >> 4;
    const _Float16* p = Bt + (size_t)col * ldb + k0 + grp * 8;
    h8 lo = *(const h8*)p;
    h8 hi = *(const h8*)(p + 16);
    return __builtin_shufflevector(lo, hi, 0,1,2,3,4,5,6,7,8,9,10,11,12,13,14,15);
}

__device__ __forceinline__ f8 wmma_f16(h16 a, h16 b, f8 c) {
    return __builtin_amdgcn_wmma_f32_16x16x32_f16(false, a, false, b,
                                                  (short)0, c, false, false);
}

// =====================================================================
// TDM: DMA one contiguous 64KB panel (8192 x 8B elements, 1-D tile) from
// global memory into LDS.  Issued by a single wave; caller must barrier.
// D# layout per CDNA5 ISA (count=1, type=2,
// data_size=8B, tile_dim0=tensor_dim0=8192, tile_dim1=1).
// 6-arg builtin form (amdgpu-toolchain / therock-10.0 headers).
// =====================================================================
__device__ __forceinline__ void tdm_load_64k_panel(const _Float16* gsrc, void* lds_dst) {
    unsigned long long ga = (unsigned long long)(size_t)gsrc;
    unsigned lo = (unsigned)(size_t)lds_dst;   // LDS byte offset (addr[31:0])
    u32x4 g0;
    g0[0] = 1u;                                 // count=1, is_restore=0, gather off
    g0[1] = lo;                                 // lds_addr
    g0[2] = (unsigned)ga;                       // global_addr[31:0]
    g0[3] = (unsigned)((ga >> 32) & 0x01FFFFFFu) | 0x80000000u; // ga[56:32] | type=2
    i32x8 g1;
    g1[0] = 0x00030000;   // workgroup_mask=0, data_size=3 (8B)
    g1[1] = 0x20000000;   // tensor_dim0[15:0]=8192 in bits[63:48]
    g1[2] = 0x00010000;   // tensor_dim1=1 in bits[95:80]
    g1[3] = 0x20000000;   // tile_dim0=8192 in bits[127:112]
    g1[4] = 0x00000001;   // tile_dim1=1
    g1[5] = 0x00002000;   // tensor_dim0_stride=8192
    g1[6] = 0x20000000;   // tensor_dim1_stride=8192 in bits[223:208]
    g1[7] = 0;
    i32x4 z4; z4[0] = 0; z4[1] = 0; z4[2] = 0; z4[3] = 0;
    i32x8 z8; z8[0] = 0; z8[1] = 0; z8[2] = 0; z8[3] = 0;
                z8[4] = 0; z8[5] = 0; z8[6] = 0; z8[7] = 0;
    __builtin_amdgcn_tensor_load_to_lds(g0, g1, z4, z4, z8, 0);
    __builtin_amdgcn_s_wait_tensorcnt(0);
}

// ---- block-wide reductions (256 threads) ----
__device__ __forceinline__ float blk_max(float v, float* sm) {
    int tid = threadIdx.x;
    sm[tid] = v; __syncthreads();
#pragma unroll
    for (int off = 128; off > 0; off >>= 1) {
        if (tid < off) sm[tid] = fmaxf(sm[tid], sm[tid + off]);
        __syncthreads();
    }
    float r = sm[0]; __syncthreads();
    return r;
}
__device__ __forceinline__ float blk_sum(float v, float* sm) {
    int tid = threadIdx.x;
    sm[tid] = v; __syncthreads();
#pragma unroll
    for (int off = 128; off > 0; off >>= 1) {
        if (tid < off) sm[tid] = sm[tid] + sm[tid + off];
        __syncthreads();
    }
    float r = sm[0]; __syncthreads();
    return r;
}

// =====================================================================
// Kernel 1: transpose + f16-convert the four E x E weight matrices.
// Wt[w][n*E + k] = (f16) W[w][k*E + n]   (N-major for the B-operand loader)
// =====================================================================
__global__ void prep_w_kernel(const float* __restrict__ Wq, const float* __restrict__ Wk,
                              const float* __restrict__ Wv, const float* __restrict__ Wo,
                              _Float16* __restrict__ Wt) {
    int i   = blockIdx.x * blockDim.x + threadIdx.x;   // < 4*E*E
    int w   = i >> 18;                                 // E*E = 262144
    int rem = i & (262144 - 1);
    int n   = rem >> 9;
    int k   = rem & 511;
    const float* W = (w == 0) ? Wq : (w == 1) ? Wk : (w == 2) ? Wv : Wo;
    Wt[((size_t)w << 18) + (size_t)n * Ec + k] = (_Float16)W[(size_t)k * Ec + n];
}

// =====================================================================
// Kernel 2: allBelow[b] = all(mask[b,:] < threshold)
// =====================================================================
__global__ void below_kernel(const float* __restrict__ mask,
                             const float* __restrict__ thr,
                             int* __restrict__ allBelow) {
    __shared__ int sm[256];
    int b = blockIdx.x, tid = threadIdx.x;
    float t = *thr;
    int all = 1;
    for (int s = tid; s < Sc; s += 256) all &= (mask[(size_t)b * Sc + s] < t) ? 1 : 0;
    sm[tid] = all; __syncthreads();
#pragma unroll
    for (int off = 128; off > 0; off >>= 1) {
        if (tid < off) sm[tid] &= sm[tid + off];
        __syncthreads();
    }
    if (tid == 0) allBelow[b] = sm[0];
}

// =====================================================================
// Kernel 3: projection GEMM  Y = (X @ W + bias) * scale  -> f16, head-split.
// X: (B*L, E) fp32.  Weight panel (64 cols x 512 k, 64KB contiguous in the
// N-major Wt) is DMAed to LDS via TDM; all 8 waves feed WMMA B from LDS.
// Block = 256 thr = 8 waves (4M x 2N); block tile 64x64; each wave 16x32.
// vmode 0: dst (B,H,L,D); 1: dst (B,H,D,S).
// =====================================================================
__global__ void __launch_bounds__(256)
proj_kernel(const float* __restrict__ X, const _Float16* __restrict__ Wt,
            const float* __restrict__ bias, float scale,
            _Float16* __restrict__ dst, int vmode) {
    __shared__ __align__(16) _Float16 sB[64 * Ec];   // 64 KB

    int lane = threadIdx.x & 31;
    int w    = threadIdx.x >> 5;
    int wm   = w & 3, wn = w >> 2;

    if (w == 0) tdm_load_64k_panel(Wt + (size_t)blockIdx.y * 64 * Ec, sB);
    __syncthreads();

    int m0  = blockIdx.x * 64 + wm * 16;
    int row = m0 + (lane & 15);
    int nl0 = wn * 32 + (lane & 15);          // local col, tile 0 (tile 1 = +16)

    f8 acc0 = F8_ZERO, acc1 = F8_ZERO;
#pragma unroll 4
    for (int k0 = 0; k0 < Ec; k0 += 32) {
        h16 a  = load_a_f32(X, Ec, row, k0, lane);
        h16 b0 = load_b_t(sB, Ec, nl0,      k0, lane);
        h16 b1 = load_b_t(sB, Ec, nl0 + 16, k0, lane);
        acc0 = wmma_f16(a, b0, acc0);
        acc1 = wmma_f16(a, b1, acc1);
    }

    int mb = m0 + (lane >> 4) * 8;
#pragma unroll
    for (int t = 0; t < 2; ++t) {
        int col = blockIdx.y * 64 + nl0 + t * 16;
        f8 acc = t ? acc1 : acc0;
        float bv = bias[col];
        int hh = col >> 6, d = col & 63;
#pragma unroll
        for (int r = 0; r < 8; ++r) {
            int m  = mb + r;
            int bb = m >> 10, l = m & 1023;
            float y = (acc[r] + bv) * scale;
            size_t idx = vmode
                ? ((size_t)(bb * Hc + hh) * Dc + d) * Sc + l      // (B,H,D,S)
                : ((size_t)(bb * Hc + hh) * Lc + l) * Dc + d;     // (B,H,L,D)
            dst[idx] = (_Float16)y;
        }
    }
}

// =====================================================================
// Kernel 4: attention logits  (per b,h):  Qh (LxD) @ Kh^T -> fp32 (L,S),
// written straight into the masked_attn_weight output region (store-bound;
// 128MB region is L2-resident, 192MB global L2).
// =====================================================================
__global__ void __launch_bounds__(256)
logits_kernel(const _Float16* __restrict__ Qh,
              const _Float16* __restrict__ Kh,
              float* __restrict__ C) {
    int bh = blockIdx.z;
    const _Float16* A  = Qh + (size_t)bh * Lc * Dc;   // lda = D
    const _Float16* Bt = Kh + (size_t)bh * Sc * Dc;   // b[j] = K[s][k]; ldb = D
    float* Cb = C + (size_t)bh * Lc * Sc;

    int lane = threadIdx.x & 31;
    int w    = threadIdx.x >> 5;
    int wm   = w & 3, wn = w >> 2;
    int m0   = blockIdx.x * 64 + wm * 16;
    int row  = m0 + (lane & 15);
    int c0   = blockIdx.y * 64 + wn * 32 + (lane & 15);

    f8 acc0 = F8_ZERO, acc1 = F8_ZERO;
#pragma unroll
    for (int k0 = 0; k0 < Dc; k0 += 32) {
        h16 a  = load_a_f16(A, Dc, row, k0, lane);
        h16 b0 = load_b_t(Bt, Dc, c0,      k0, lane);
        h16 b1 = load_b_t(Bt, Dc, c0 + 16, k0, lane);
        acc0 = wmma_f16(a, b0, acc0);
        acc1 = wmma_f16(a, b1, acc1);
    }

    int mb = m0 + (lane >> 4) * 8;
#pragma unroll
    for (int r = 0; r < 8; ++r) {
        Cb[(size_t)(mb + r) * Sc + c0]      = acc0[r];
        Cb[(size_t)(mb + r) * Sc + c0 + 16] = acc1[r];
    }
}

// =====================================================================
// Kernel 5: out_attn_weight = mean over heads of softmax(logits).
// One block per (b,l); reads all 8 head rows (L2-resident).
// =====================================================================
__global__ void softmax_mean_kernel(const float* __restrict__ logits,
                                    float* __restrict__ out2) {
    __shared__ float sm[256];
    int bl = blockIdx.x, b = bl >> 10, l = bl & 1023;
    int tid = threadIdx.x;
    float acc[4] = {0.f, 0.f, 0.f, 0.f};
    for (int hh = 0; hh < Hc; ++hh) {
        const float* row = logits + ((size_t)(b * Hc + hh) * Lc + l) * Sc;
        float v[4];
        float mx = -__builtin_inff();
#pragma unroll
        for (int i = 0; i < 4; ++i) { v[i] = row[tid + i * 256]; mx = fmaxf(mx, v[i]); }
        mx = blk_max(mx, sm);
        float se = 0.f;
#pragma unroll
        for (int i = 0; i < 4; ++i) { v[i] = __expf(v[i] - mx); se += v[i]; }
        se = blk_sum(se, sm);
        float inv = 1.f / se;
#pragma unroll
        for (int i = 0; i < 4; ++i) acc[i] += v[i] * inv * (1.f / Hc);
    }
    float* o = out2 + (size_t)bl * Sc;
#pragma unroll
    for (int i = 0; i < 4; ++i) o[tid + i * 256] = acc[i];
}

// =====================================================================
// Kernel 6: masked softmax, IN PLACE over the (B,H,L,S) output region.
// =====================================================================
__global__ void masked_softmax_kernel(float* __restrict__ logits,
                                      const float* __restrict__ mask,
                                      const float* __restrict__ thr,
                                      const int* __restrict__ allBelow) {
    __shared__ float sm[256];
    size_t bhl = blockIdx.x;                 // < B*H*L
    int b = (int)(bhl >> 13);                // H*L = 8192
    int tid = threadIdx.x;
    float t = *thr;
    int apply = !allBelow[b];
    float* row = logits + bhl * Sc;
    const float* mrow = mask + (size_t)b * Sc;

    float v[4];
    float mx = -__builtin_inff();
#pragma unroll
    for (int i = 0; i < 4; ++i) {
        int s = tid + i * 256;
        float lv = row[s];
        if (apply && (mrow[s] < t)) lv = -__builtin_inff();
        v[i] = lv;
        mx = fmaxf(mx, lv);
    }
    mx = blk_max(mx, sm);
    float se = 0.f;
#pragma unroll
    for (int i = 0; i < 4; ++i) { v[i] = __expf(v[i] - mx); se += v[i]; }
    se = blk_sum(se, sm);
    float inv = 1.f / se;
#pragma unroll
    for (int i = 0; i < 4; ++i) row[tid + i * 256] = v[i] * inv;
}

// =====================================================================
// Kernel 7: context  x_h = P (LxS, fp32 -> f16 on the fly) @ V (Vt is
// (B,H,D,S) so the B-operand per lane is contiguous in K=s).  Prefetch of
// the next A K-slab lowers to global_prefetch_b8.  Result f16 (B,L,E).
// =====================================================================
__global__ void __launch_bounds__(256)
pv_kernel(const float* __restrict__ P,
          const _Float16* __restrict__ Vt,
          _Float16* __restrict__ xh) {
    int bh = blockIdx.z, bb = bh >> 3, hh = bh & 7;
    const float*    A  = P  + (size_t)bh * Lc * Sc;   // lda = S
    const _Float16* Bt = Vt + (size_t)bh * Dc * Sc;   // ldb = S

    int lane = threadIdx.x & 31;
    int w    = threadIdx.x >> 5;
    int wm   = w & 3, wn = w >> 2;
    int m0   = blockIdx.x * 64 + wm * 16;
    int row  = m0 + (lane & 15);
    int c0   = wn * 32 + (lane & 15);                 // d-index, tile 1 = +16

    const float* Arow = A + (size_t)row * Sc;

    f8 acc0 = F8_ZERO, acc1 = F8_ZERO;
    for (int k0 = 0; k0 < Sc; k0 += 32) {
        if (k0 + 256 < Sc)
            __builtin_prefetch((const void*)(Arow + k0 + 256), 0, 1);
        h16 a  = load_a_f32(A, Sc, row, k0, lane);
        h16 b0 = load_b_t(Bt, Sc, c0,      k0, lane);
        h16 b1 = load_b_t(Bt, Sc, c0 + 16, k0, lane);
        acc0 = wmma_f16(a, b0, acc0);
        acc1 = wmma_f16(a, b1, acc1);
    }

    int mb = m0 + (lane >> 4) * 8;
#pragma unroll
    for (int r = 0; r < 8; ++r) {
        int m = mb + r;
        size_t base = ((size_t)(bb * Lc + m)) * Ec + hh * Dc;
        xh[base + c0]      = (_Float16)acc0[r];
        xh[base + c0 + 16] = (_Float16)acc1[r];
    }
}

// =====================================================================
// Kernel 8: output projection  x = xh (4096x512 f16) @ Wo + bo -> fp32.
// Same TDM-staged weight panel scheme as proj_kernel.
// =====================================================================
__global__ void __launch_bounds__(256)
oproj_kernel(const _Float16* __restrict__ Xh,
             const _Float16* __restrict__ Wt,
             const float* __restrict__ bias,
             float* __restrict__ out) {
    __shared__ __align__(16) _Float16 sB[64 * Ec];   // 64 KB

    int lane = threadIdx.x & 31;
    int w    = threadIdx.x >> 5;
    int wm   = w & 3, wn = w >> 2;

    if (w == 0) tdm_load_64k_panel(Wt + (size_t)blockIdx.y * 64 * Ec, sB);
    __syncthreads();

    int m0  = blockIdx.x * 64 + wm * 16;
    int row = m0 + (lane & 15);
    int nl0 = wn * 32 + (lane & 15);

    f8 acc0 = F8_ZERO, acc1 = F8_ZERO;
#pragma unroll 4
    for (int k0 = 0; k0 < Ec; k0 += 32) {
        h16 a  = load_a_f16(Xh, Ec, row, k0, lane);
        h16 b0 = load_b_t(sB, Ec, nl0,      k0, lane);
        h16 b1 = load_b_t(sB, Ec, nl0 + 16, k0, lane);
        acc0 = wmma_f16(a, b0, acc0);
        acc1 = wmma_f16(a, b1, acc1);
    }

    int mb = m0 + (lane >> 4) * 8;
#pragma unroll
    for (int t = 0; t < 2; ++t) {
        int col = blockIdx.y * 64 + nl0 + t * 16;
        f8 acc = t ? acc1 : acc0;
        float bv = bias[col];
#pragma unroll
        for (int r = 0; r < 8; ++r)
            out[(size_t)(mb + r) * Ec + col] = acc[r] + bv;
    }
}

// =====================================================================
// Launch pipeline
// =====================================================================
extern "C" void kernel_launch(void* const* d_in, const int* in_sizes, int n_in,
                              void* d_out, int out_size, void* d_ws, size_t ws_size,
                              hipStream_t stream) {
    (void)in_sizes; (void)n_in; (void)out_size; (void)ws_size;

    const float* q    = (const float*)d_in[0];
    const float* k    = (const float*)d_in[1];
    const float* v    = (const float*)d_in[2];
    const float* mask = (const float*)d_in[3];
    const float* thr  = (const float*)d_in[4];
    const float* Wq = (const float*)d_in[5];  const float* bq = (const float*)d_in[6];
    const float* Wk = (const float*)d_in[7];  const float* bk = (const float*)d_in[8];
    const float* Wv = (const float*)d_in[9];  const float* bv = (const float*)d_in[10];
    const float* Wo = (const float*)d_in[11]; const float* bo = (const float*)d_in[12];

    // workspace layout (bytes)
    char* ws = (char*)d_ws;
    _Float16* Wt = (_Float16*)(ws + 0);           // 4 * E*E f16        (2 MB)
    _Float16* Qh = (_Float16*)(ws + 2097152);     // (B,H,L,D) f16      (4 MB)
    _Float16* Kh = (_Float16*)(ws + 6291456);     // (B,H,S,D) f16      (4 MB)
    _Float16* Vt = (_Float16*)(ws + 10485760);    // (B,H,D,S) f16      (4 MB)
    _Float16* Xh = (_Float16*)(ws + 14680064);    // (B,L,E)   f16      (4 MB)
    int*   flags = (int*)     (ws + 18874368);    // B ints

    // output layout (fp32, concatenated in return order)
    float* outX   = (float*)d_out;                        // (B,L,E)
    float* outAW  = outX  + (size_t)Bc * Lc * Ec;         // (B,L,S)
    float* outMAW = outAW + (size_t)Bc * Lc * Sc;         // (B,H,L,S)

    const float scale = 0.125f;   // D^-0.5, D = 64

    prep_w_kernel<<<4096, 256, 0, stream>>>(Wq, Wk, Wv, Wo, Wt);
    below_kernel<<<Bc, 256, 0, stream>>>(mask, thr, flags);

    proj_kernel<<<dim3(64, 8), 256, 0, stream>>>(q, Wt,              bq, scale, Qh, 0);
    proj_kernel<<<dim3(64, 8), 256, 0, stream>>>(k, Wt + 262144,     bk, 1.0f,  Kh, 0);
    proj_kernel<<<dim3(64, 8), 256, 0, stream>>>(v, Wt + 2 * 262144, bv, 1.0f,  Vt, 1);

    logits_kernel<<<dim3(16, 16, Bc * Hc), 256, 0, stream>>>(Qh, Kh, outMAW);

    softmax_mean_kernel<<<Bc * Lc, 256, 0, stream>>>(outMAW, outAW);
    masked_softmax_kernel<<<Bc * Hc * Lc, 256, 0, stream>>>(outMAW, mask, thr, flags);

    pv_kernel<<<dim3(16, 1, Bc * Hc), 256, 0, stream>>>(outMAW, Vt, Xh);
    oproj_kernel<<<dim3(64, 8), 256, 0, stream>>>(Xh, Wt + 3 * 262144, bo, outX);
}
